// GRU_1340029796816
// MI455X (gfx1250) — matrix operand order repaired and checked by reading, not compiled, not run on hardware
//
#include <hip/hip_runtime.h>

typedef __attribute__((ext_vector_type(16))) _Float16 v16h;
typedef __attribute__((ext_vector_type(8)))  float    v8f;

#define TSTEPS 256
#define NB     4096
#define BT     16

// LDS layout (bytes)
#define OUT1_OFF   0            // [T][16][16] f16  = 131072
#define H1_OFF     131072       // [16][16]   f16  = 512
#define HB_OFF     131584       // [2][16][64] f16 = 4096
#define HSUM_OFF   135680       // [2][16][64] f32 = 8192
#define SMEM_BYTES 143872

#define LOG2E 1.4426950408889634f

// Branchless transcendentals on the hardware TRANS pipes (no libm, no EXEC branches).
__device__ __forceinline__ float fast_sigmoid(float x) {
    return __builtin_amdgcn_rcpf(1.0f + __builtin_amdgcn_exp2f(x * (-LOG2E)));
}
__device__ __forceinline__ float fast_tanh(float x) {
#if __has_builtin(__builtin_amdgcn_tanhf)
    return __builtin_amdgcn_tanhf(x);          // v_tanh_f32 (CDNA5 TRANS op)
#else
    // tanh(x) = 2*sigmoid(2x) - 1
    return 2.0f * __builtin_amdgcn_rcpf(1.0f + __builtin_amdgcn_exp2f(x * (-2.0f * LOG2E))) - 1.0f;
#endif
}

__device__ __forceinline__ v8f wmma_f16(v16h a, v16h b, v8f c) {
    // D = A(16x32 f16) * B(32x16 f16) + C(16x16 f32)
    return __builtin_amdgcn_wmma_f32_16x16x32_f16(false, a, false, b, (short)0, c, false, false);
}

// Build one B-matrix tile (32x16, f16) from row-major f32 weights W[ncols][ldw].
// B element: lane -> N = lane%16 (weight row col0+N), K = 16*(lane/16) + e (weight col).
// Zero-pads K >= kmax.
__device__ __forceinline__ v16h load_B(const float* W, int ldw, int col0, int k0, int kmax, int lane) {
    int n  = col0 + (lane & 15);
    int kb = k0 + ((lane >> 4) << 4);
    v16h b;
#pragma unroll
    for (int e = 0; e < 16; ++e) {
        int k = kb + e;
        b[e] = (k < kmax) ? (_Float16)W[(size_t)n * ldw + k] : (_Float16)0.0f;
    }
    return b;
}

// Build one A-matrix tile (16x32, f16) from row-major f16 LDS S[rows][rowstride].
// A element: row = lane%16; K = 16*half+e.
// Zero-pads relative K >= valid.
__device__ __forceinline__ v16h load_A(const _Float16* S, int rowstride, int row, int j0, int valid, int lane) {
    int half_ = (lane >> 4) & 1;
    v16h a;
#pragma unroll
    for (int e = 0; e < 16; ++e) {
        int rel = 16 * half_ + e;
        a[e] = (rel < valid) ? S[(size_t)row * rowstride + j0 + rel] : (_Float16)0.0f;
    }
    return a;
}

__global__ __launch_bounds__(64, 1)
void gru_bidir_fused(const float* __restrict__ X,
                     const float* __restrict__ w_ih1, const float* __restrict__ w_hh1,
                     const float* __restrict__ b_ih1, const float* __restrict__ b_hh1,
                     const float* __restrict__ w_ih2f, const float* __restrict__ w_hh2f,
                     const float* __restrict__ b_ih2f, const float* __restrict__ b_hh2f,
                     const float* __restrict__ w_ih2b, const float* __restrict__ w_hh2b,
                     const float* __restrict__ b_ih2b, const float* __restrict__ b_hh2b,
                     const float* __restrict__ W1, const float* __restrict__ b1,
                     const float* __restrict__ W2, const float* __restrict__ b2,
                     float* __restrict__ out)
{
    extern __shared__ char smem[];
    _Float16* out1  = (_Float16*)(smem + OUT1_OFF);   // [t][m][j] f16
    _Float16* h1buf = (_Float16*)(smem + H1_OFF);     // [m][j] f16
    _Float16* hbuf  = (_Float16*)(smem + HB_OFF);     // [wave][m][j] f16
    float*    hsum  = (float*)(smem + HSUM_OFF);      // [wave][m][j] f32

    const int tid   = threadIdx.x;
    const int wave  = tid >> 5;
    const int lane  = tid & 31;
    const int half_ = (lane >> 4) & 1;
    const int col   = lane & 15;
    const int b0    = blockIdx.x * BT;

    // ================= Layer 1 (H=16), wave 0 only =================
    if (wave == 0) {
        v16h B1[3];
#pragma unroll
        for (int nt = 0; nt < 3; ++nt) B1[nt] = load_B(w_hh1, 16, 16 * nt, 0, 16, lane);

        float wi0[3], wi1[3], brz[2];
#pragma unroll
        for (int nt = 0; nt < 3; ++nt) {
            int g = 16 * nt + col;
            wi0[nt] = w_ih1[g * 2 + 0];
            wi1[nt] = w_ih1[g * 2 + 1];
            if (nt < 2) brz[nt] = b_ih1[g] + b_hh1[g];
        }
        const float bihn = b_ih1[32 + col];
        const float bhhn = b_hh1[32 + col];

        v8f h = {};
#pragma clang loop unroll(disable)
        for (int t = 0; t < TSTEPS; ++t) {
            // stash h (C layout -> row major f16) for A-layout pickup
#pragma unroll
            for (int v = 0; v < 8; ++v)
                h1buf[(v + 8 * half_) * 16 + col] = (_Float16)h[v];
            asm volatile("s_wait_dscnt 0x0" ::: "memory");
            v16h a = load_A(h1buf, 16, col, 0, 16, lane);

            // per-lane x for the 8 batch rows this lane owns in the C layout
            float x0[8], x1[8];
#pragma unroll
            for (int v = 0; v < 8; ++v) {
                int m = v + 8 * half_;
                const float* xp = X + ((size_t)(b0 + m) * TSTEPS + t) * 2;
                x0[v] = xp[0];
                x1[v] = xp[1];
            }
            v8f cr, cz, cn;
#pragma unroll
            for (int v = 0; v < 8; ++v) {
                cr[v] = brz[0] + wi0[0] * x0[v] + wi1[0] * x1[v];
                cz[v] = brz[1] + wi0[1] * x0[v] + wi1[1] * x1[v];
                cn[v] = bhhn;
            }
            cr = wmma_f16(a, B1[0], cr);
            cz = wmma_f16(a, B1[1], cz);
            cn = wmma_f16(a, B1[2], cn);
#pragma unroll
            for (int v = 0; v < 8; ++v) {
                float r  = fast_sigmoid(cr[v]);
                float z  = fast_sigmoid(cz[v]);
                float xn = bihn + wi0[2] * x0[v] + wi1[2] * x1[v];
                float n  = fast_tanh(xn + r * cn[v]);
                h[v] = (1.0f - z) * n + z * h[v];
                out1[((size_t)t * 16 + (v + 8 * half_)) * 16 + col] = (_Float16)h[v];
            }
        }
    }
    __syncthreads();

    // ============ Layer 2 (H=64): wave0 = forward, wave1 = backward ============
    {
        const float* Whh = wave ? w_hh2b : w_hh2f;
        const float* Wih = wave ? w_ih2b : w_ih2f;
        const float* Bih = wave ? b_ih2b : b_ih2f;
        const float* Bhh = wave ? b_hh2b : b_hh2f;
        _Float16* hb = hbuf + wave * (16 * 64);

        v16h Bh[2][12], Bx[12];
#pragma unroll
        for (int nt = 0; nt < 12; ++nt) {
            Bh[0][nt] = load_B(Whh, 64, 16 * nt, 0, 64, lane);
            Bh[1][nt] = load_B(Whh, 64, 16 * nt, 32, 64, lane);
            Bx[nt]    = load_B(Wih, 16, 16 * nt, 0, 16, lane);
            // Fold biases into the spare K=16 row of the Bx tile (pairs with A_x K=16 == 1.0):
            //   r/z tiles (nt<8): b_ih + b_hh ; xn tiles (nt>=8): b_ih only.
            float bias = (nt < 8) ? (Bih[16 * nt + col] + Bhh[16 * nt + col])
                                  : Bih[16 * nt + col];
            if (half_) Bx[nt][0] = (_Float16)bias;   // lanes 16..31, e=0 -> K=16
        }
        float bhn[4];
#pragma unroll
        for (int k = 0; k < 4; ++k) bhn[k] = Bhh[128 + 16 * k + col];

        v8f h[4] = {};
#pragma clang loop unroll(disable)
        for (int t = 0; t < TSTEPS; ++t) {
            int tt = wave ? (TSTEPS - 1 - t) : t;
            // stash h (C layout) as f16 row-major for A pickup
#pragma unroll
            for (int k = 0; k < 4; ++k)
#pragma unroll
                for (int v = 0; v < 8; ++v)
                    hb[(v + 8 * half_) * 64 + 16 * k + col] = (_Float16)h[k][v];
            asm volatile("s_wait_dscnt 0x0" ::: "memory");
            v16h a0 = load_A(hb, 64, col, 0, 32, lane);
            v16h a1 = load_A(hb, 64, col, 32, 32, lane);
            v16h ax = load_A(out1 + (size_t)tt * 256, 16, col, 0, 16, lane);
            if (half_) ax[0] = (_Float16)1.0f;   // K=16 constant-1 column (bias row)

            const v8f zero = {};
            v8f acc[12], accx[4];
#pragma unroll
            for (int nt = 0; nt < 8; ++nt) {        // r (0..3) and z (4..7) gate tiles
                acc[nt] = wmma_f16(ax, Bx[nt], zero);        // x-proj + (b_ih+b_hh)
                acc[nt] = wmma_f16(a0, Bh[0][nt], acc[nt]);
                acc[nt] = wmma_f16(a1, Bh[1][nt], acc[nt]);
            }
#pragma unroll
            for (int k = 0; k < 4; ++k) {           // n gate: keep hn and xn separate
                int nt = 8 + k;
                accx[k] = wmma_f16(ax, Bx[nt], zero);        // xn + b_ih
#pragma unroll
                for (int v = 0; v < 8; ++v) acc[nt][v] = bhn[k];
                acc[nt] = wmma_f16(a0, Bh[0][nt], acc[nt]);  // hn + b_hh
                acc[nt] = wmma_f16(a1, Bh[1][nt], acc[nt]);
            }
#pragma unroll
            for (int k = 0; k < 4; ++k)
#pragma unroll
                for (int v = 0; v < 8; ++v) {
                    float r = fast_sigmoid(acc[k][v]);
                    float z = fast_sigmoid(acc[4 + k][v]);
                    float n = fast_tanh(accx[k][v] + r * acc[8 + k][v]);
                    h[k][v] = (1.0f - z) * n + z * h[k][v];
                }
        }
        // final hidden state of this direction -> LDS (f32)
#pragma unroll
        for (int k = 0; k < 4; ++k)
#pragma unroll
            for (int v = 0; v < 8; ++v)
                hsum[wave * (16 * 64) + (v + 8 * half_) * 64 + 16 * k + col] = h[k][v];
    }
    __syncthreads();

    // ================= MLP (h_fwd + h_bwd) -> 32 -> 8, lanes 0..15 =================
    if (tid < 16) {
        int row = tid;
        float hrow[64];
#pragma unroll
        for (int j = 0; j < 64; ++j)
            hrow[j] = hsum[row * 64 + j] + hsum[16 * 64 + row * 64 + j];
        float z[8];
#pragma unroll
        for (int o = 0; o < 8; ++o) z[o] = b2[o];
#pragma clang loop unroll(disable)
        for (int m = 0; m < 32; ++m) {
            float s = b1[m];
#pragma unroll
            for (int j = 0; j < 64; ++j) s += hrow[j] * W1[m * 64 + j];
#pragma unroll
            for (int o = 0; o < 8; ++o) z[o] += s * W2[o * 32 + m];
        }
#pragma unroll
        for (int o = 0; o < 8; ++o)
            out[(size_t)(b0 + row) * 8 + o] = z[o];
    }
}

extern "C" void kernel_launch(void* const* d_in, const int* in_sizes, int n_in,
                              void* d_out, int out_size, void* d_ws, size_t ws_size,
                              hipStream_t stream) {
    (void)in_sizes; (void)n_in; (void)out_size; (void)d_ws; (void)ws_size;
    const float* X      = (const float*)d_in[0];
    const float* w_ih1  = (const float*)d_in[1];
    const float* w_hh1  = (const float*)d_in[2];
    const float* b_ih1  = (const float*)d_in[3];
    const float* b_hh1  = (const float*)d_in[4];
    const float* w_ih2f = (const float*)d_in[5];
    const float* w_hh2f = (const float*)d_in[6];
    const float* b_ih2f = (const float*)d_in[7];
    const float* b_hh2f = (const float*)d_in[8];
    const float* w_ih2b = (const float*)d_in[9];
    const float* w_hh2b = (const float*)d_in[10];
    const float* b_ih2b = (const float*)d_in[11];
    const float* b_hh2b = (const float*)d_in[12];
    const float* W1     = (const float*)d_in[13];
    const float* b1     = (const float*)d_in[14];
    const float* W2     = (const float*)d_in[15];
    const float* b2     = (const float*)d_in[16];
    float* out = (float*)d_out;

    hipFuncSetAttribute((const void*)gru_bidir_fused,
                        hipFuncAttributeMaxDynamicSharedMemorySize, SMEM_BYTES);

    dim3 grid(NB / BT), block(64);
    gru_bidir_fused<<<grid, block, SMEM_BYTES, stream>>>(
        X, w_ih1, w_hh1, b_ih1, b_hh1,
        w_ih2f, w_hh2f, b_ih2f, b_hh2f,
        w_ih2b, w_hh2b, b_ih2b, b_hh2b,
        W1, b1, W2, b2, out);
}
